// LinearCrossEntropyLoss_25409026523316
// MI455X (gfx1250) — compile-verified, matrix operand order checked by reading, not run on hardware
//
#include <hip/hip_runtime.h>
#include <hip/hip_bf16.h>
#include <math.h>

// ---------------------------------------------------------------------------
// Fused Linear + CrossEntropy for MI455X (gfx1250, wave32, WMMA).
//   hidden [B=2, S=4096, D=4096] fp32   -> T = 8192 token rows
//   weight [V=32000, D=4096]     fp32
//   targets[T] int32 (JAX default x64-off => int32)
// logits = hidden @ weight^T fused with logsumexp; never materialized to HBM.
// GEMM on v_wmma_f32_16x16x32_bf16 with hi/lo bf16 split (3 chained WMMAs
// per tile => ~fp32-class accuracy at bf16-pipe rate). 2-deep software
// pipeline: fp32 tile k+1 staged in VGPRs + double-buffered LDS planes so
// global loads overlap the WMMAs of tile k.
// ---------------------------------------------------------------------------

typedef __attribute__((ext_vector_type(16))) __bf16 v16bf;
typedef __attribute__((ext_vector_type(8)))  __bf16 v8bf;
typedef __attribute__((ext_vector_type(8)))  float  v8f;

#define T_TOK   8192
#define DIM     4096
#define VOCAB   32000
#define TM      64          // tokens per workgroup
#define TN      128         // vocab cols per workgroup
#define KB      32          // K block (bf16 WMMA K)
#define NVS     (VOCAB/TN)  // 250 vocab slices
#define IGNORE_INDEX (-100)

// LDS arena: double-buffered staging planes aliased with the epilogue logits.
//   buffer p at p*24576: Ahi 4KB | Alo 4KB | Bhi 8KB | Blo 8KB   (2x24KB)
//   epilogue:            logits 32KB | redm 1KB | reds 1KB       (<=48KB)
#define STAGE_BYTES 24576
#define ARENA_BYTES 49152

// workspace layout, in floats
#define WS_PMAX  ((size_t)0)
#define WS_PSUM  (WS_PMAX + (size_t)T_TOK*NVS)
#define WS_TGT   (WS_PSUM + (size_t)T_TOK*NVS)
#define WS_BSUM  (WS_TGT  + (size_t)T_TOK)
#define WS_BCNT  (WS_BSUM + 64)

static __device__ __forceinline__ void split_store(__bf16* hi, __bf16* lo, float x) {
    __bf16 h = (__bf16)x;                 // RNE to bf16
    *hi = h;
    *lo = (__bf16)(x - (float)h);         // residual
}

static __device__ __forceinline__ void split_store4(__bf16* hi, __bf16* lo, float4 x) {
    split_store(hi+0, lo+0, x.x); split_store(hi+1, lo+1, x.y);
    split_store(hi+2, lo+2, x.z); split_store(hi+3, lo+3, x.w);
}

__global__ __launch_bounds__(256)
void ce_partials_kernel(const float* __restrict__ hidden,
                        const float* __restrict__ weight,
                        const int*   __restrict__ targets,
                        float*       __restrict__ ws)
{
    __shared__ __align__(16) unsigned char arena[ARENA_BYTES];

    const int tid  = threadIdx.x;
    const int lane = tid & 31;
    const int wave = tid >> 5;           // 8 waves
    const int mW   = wave & 1;           // 2 row bands of 32
    const int nW   = wave >> 1;          // 4 col bands of 32

    const int t0 = blockIdx.x * TM;      // token base (x fastest: weight tile stays in L2)
    const int v0 = blockIdx.y * TN;      // vocab base

    // Fixed per-thread staging coordinates.
    const int aflat = tid * 8,  ar = aflat >> 5, ak = aflat & 31;   // A: 8 floats/thread
    const int bflat = tid * 16, br = bflat >> 5, bk = bflat & 31;   // B: 16 floats/thread
    const float* aSrc = hidden + (size_t)(t0 + ar) * DIM + ak;
    const float* bSrc = weight + (size_t)(v0 + br) * DIM + bk;

    v8f acc[2][2];
    #pragma unroll
    for (int i = 0; i < 2; ++i)
        #pragma unroll
        for (int j = 0; j < 2; ++j)
            #pragma unroll
            for (int e = 0; e < 8; ++e) acc[i][j][e] = 0.0f;

    // ---- prologue: stage tile k=0 into registers ----
    float4 aR0 = *(const float4*)(aSrc);
    float4 aR1 = *(const float4*)(aSrc + 4);
    float4 bR0 = *(const float4*)(bSrc);
    float4 bR1 = *(const float4*)(bSrc + 4);
    float4 bR2 = *(const float4*)(bSrc + 8);
    float4 bR3 = *(const float4*)(bSrc + 12);

    int p = 0;
    for (int k0 = 0; k0 < DIM; k0 += KB) {
        unsigned char* buf = arena + p * STAGE_BYTES;
        __bf16* lAhi = (__bf16*)(buf);
        __bf16* lAlo = (__bf16*)(buf + 4096);
        __bf16* lBhi = (__bf16*)(buf + 8192);
        __bf16* lBlo = (__bf16*)(buf + 16384);

        // ---- convert current tile regs -> bf16 hi/lo planes in LDS buf[p] ----
        split_store4(&lAhi[ar*KB + ak],     &lAlo[ar*KB + ak],     aR0);
        split_store4(&lAhi[ar*KB + ak + 4], &lAlo[ar*KB + ak + 4], aR1);
        split_store4(&lBhi[br*KB + bk],      &lBlo[br*KB + bk],      bR0);
        split_store4(&lBhi[br*KB + bk + 4],  &lBlo[br*KB + bk + 4],  bR1);
        split_store4(&lBhi[br*KB + bk + 8],  &lBlo[br*KB + bk + 8],  bR2);
        split_store4(&lBhi[br*KB + bk + 12], &lBlo[br*KB + bk + 12], bR3);
        __syncthreads();

        // ---- issue global loads for tile k+1 (overlaps the WMMAs below) ----
        if (k0 + KB < DIM) {
            const float* an = aSrc + k0 + KB;
            const float* bn = bSrc + k0 + KB;
            __builtin_prefetch(an + KB, 0, 0);   // warm L2 two tiles ahead
            __builtin_prefetch(bn + KB, 0, 0);
            aR0 = *(const float4*)(an);
            aR1 = *(const float4*)(an + 4);
            bR0 = *(const float4*)(bn);
            bR1 = *(const float4*)(bn + 4);
            bR2 = *(const float4*)(bn + 8);
            bR3 = *(const float4*)(bn + 12);
        }

        // ---- fragment loads (ISA 16-bit A layout: lane<16 -> K{0..7,16..23},
        //      lane>=16 -> K{8..15,24..31}; two ds_load_b128 per 8-VGPR half) ----
        const int khalf = lane >> 4;
        const int rl    = lane & 15;
        v16bf aHi[2], aLo[2], bHi[2], bLo[2];
        #pragma unroll
        for (int mt = 0; mt < 2; ++mt) {
            const int o = (mW*32 + mt*16 + rl) * KB + khalf*8;
            ((v8bf*)&aHi[mt])[0] = *(const v8bf*)&lAhi[o];
            ((v8bf*)&aHi[mt])[1] = *(const v8bf*)&lAhi[o + 16];
            ((v8bf*)&aLo[mt])[0] = *(const v8bf*)&lAlo[o];
            ((v8bf*)&aLo[mt])[1] = *(const v8bf*)&lAlo[o + 16];
        }
        #pragma unroll
        for (int nt = 0; nt < 2; ++nt) {
            const int o = (nW*32 + nt*16 + rl) * KB + khalf*8;
            ((v8bf*)&bHi[nt])[0] = *(const v8bf*)&lBhi[o];
            ((v8bf*)&bHi[nt])[1] = *(const v8bf*)&lBhi[o + 16];
            ((v8bf*)&bLo[nt])[0] = *(const v8bf*)&lBlo[o];
            ((v8bf*)&bLo[nt])[1] = *(const v8bf*)&lBlo[o + 16];
        }

        // ---- 3-term hi/lo WMMA accumulation: hi*hi + hi*lo + lo*hi ----
        #pragma unroll
        for (int mt = 0; mt < 2; ++mt)
            #pragma unroll
            for (int nt = 0; nt < 2; ++nt) {
                acc[mt][nt] = __builtin_amdgcn_wmma_f32_16x16x32_bf16(
                    false, aHi[mt], false, bHi[nt], (short)0, acc[mt][nt], false, false);
                acc[mt][nt] = __builtin_amdgcn_wmma_f32_16x16x32_bf16(
                    false, aHi[mt], false, bLo[nt], (short)0, acc[mt][nt], false, false);
                acc[mt][nt] = __builtin_amdgcn_wmma_f32_16x16x32_bf16(
                    false, aLo[mt], false, bHi[nt], (short)0, acc[mt][nt], false, false);
            }
        p ^= 1;   // writes next iter go to the other buffer; one barrier/iter suffices
    }

    // staging region about to be reused as logits: make sure all waves are done
    __syncthreads();

    float* logits = (float*)(arena);            // 32 KB
    float* redm   = (float*)(arena + 32768);    // 1 KB
    float* reds   = (float*)(arena + 33792);    // 1 KB

    // ---- epilogue: logits tile -> LDS (C layout: vgpr r, lane -> M=r+8*(l>>4), N=l&15) ----
    {
        const int rl    = lane & 15;
        const int rhalf = lane >> 4;
        #pragma unroll
        for (int mt = 0; mt < 2; ++mt)
            #pragma unroll
            for (int nt = 0; nt < 2; ++nt) {
                const int col = nW*32 + nt*16 + rl;
                #pragma unroll
                for (int r = 0; r < 8; ++r) {
                    const int row = mW*32 + mt*16 + r + 8*rhalf;
                    logits[row*TN + col] = acc[mt][nt][r];
                }
            }
    }
    __syncthreads();

    // ---- per-row partial max / sum-exp over this 128-wide vocab slice ----
    const int row = tid >> 2;
    const int seg = tid & 3;
    const float* lrow = &logits[row*TN + seg*32];
    {
        float m = -__builtin_inff();
        #pragma unroll 8
        for (int j = 0; j < 32; ++j) m = fmaxf(m, lrow[j]);
        redm[row*4 + seg] = m;
    }
    __syncthreads();
    const float m = fmaxf(fmaxf(redm[row*4+0], redm[row*4+1]),
                          fmaxf(redm[row*4+2], redm[row*4+3]));
    {
        float s = 0.0f;
        #pragma unroll 8
        for (int j = 0; j < 32; ++j) s += __expf(lrow[j] - m);
        reds[row*4 + seg] = s;
    }
    __syncthreads();
    if (seg == 0) {
        const float s = reds[row*4+0] + reds[row*4+1] + reds[row*4+2] + reds[row*4+3];
        const int t = t0 + row;
        ws[WS_PMAX + (size_t)t*NVS + blockIdx.y] = m;
        ws[WS_PSUM + (size_t)t*NVS + blockIdx.y] = s;
        const int tgt = targets[t];
        if (tgt >= v0 && tgt < v0 + TN)          // exactly one slice owns the target
            ws[WS_TGT + t] = logits[row*TN + (tgt - v0)];
    }
}

__global__ __launch_bounds__(256)
void ce_init_kernel(float* __restrict__ ws)
{
    const int t = blockIdx.x * 256 + threadIdx.x;
    if (t < T_TOK) ws[WS_TGT + t] = 0.0f;
    if (blockIdx.x == 0 && threadIdx.x < 64) {
        ws[WS_BSUM + threadIdx.x] = 0.0f;
        ws[WS_BCNT + threadIdx.x] = 0.0f;
    }
}

__global__ __launch_bounds__(256)
void ce_reduce_kernel(const int* __restrict__ targets, float* __restrict__ ws)
{
    __shared__ float ssum[256];
    __shared__ float scnt[256];
    const int tid = threadIdx.x;
    const int t   = blockIdx.x * 256 + tid;

    float per = 0.0f, cnt = 0.0f;
    if (t < T_TOK) {
        const float* pm = &ws[WS_PMAX + (size_t)t*NVS];
        const float* ps = &ws[WS_PSUM + (size_t)t*NVS];
        float m = -__builtin_inff();
        for (int i = 0; i < NVS; ++i) m = fmaxf(m, pm[i]);
        float s = 0.0f;
        for (int i = 0; i < NVS; ++i) s += ps[i] * __expf(pm[i] - m);
        const int tgt = targets[t];
        if (tgt != IGNORE_INDEX) {
            per = m + __logf(s) - ws[WS_TGT + t];
            cnt = 1.0f;
        }
    }
    ssum[tid] = per; scnt[tid] = cnt;
    __syncthreads();
    for (int off = 128; off > 0; off >>= 1) {
        if (tid < off) { ssum[tid] += ssum[tid + off]; scnt[tid] += scnt[tid + off]; }
        __syncthreads();
    }
    if (tid == 0) {
        ws[WS_BSUM + blockIdx.x] = ssum[0];
        ws[WS_BCNT + blockIdx.x] = scnt[0];
    }
}

__global__ void ce_finalize_kernel(const float* __restrict__ ws, float* __restrict__ out)
{
    float L = 0.0f, C = 0.0f;
    for (int i = 0; i < T_TOK/256; ++i) { L += ws[WS_BSUM + i]; C += ws[WS_BCNT + i]; }
    out[0] = (C == 0.0f) ? L : (L / C);
}

extern "C" void kernel_launch(void* const* d_in, const int* in_sizes, int n_in,
                              void* d_out, int out_size, void* d_ws, size_t ws_size,
                              hipStream_t stream) {
    (void)in_sizes; (void)n_in; (void)out_size; (void)ws_size;
    const float* hidden  = (const float*)d_in[0];
    const float* weight  = (const float*)d_in[1];
    const int*   targets = (const int*)d_in[2];
    float*       out     = (float*)d_out;
    float*       ws      = (float*)d_ws;

    ce_init_kernel<<<dim3(T_TOK/256), dim3(256), 0, stream>>>(ws);
    ce_partials_kernel<<<dim3(T_TOK/TM, VOCAB/TN), dim3(256), 0, stream>>>(
        hidden, weight, targets, ws);
    ce_reduce_kernel<<<dim3(T_TOK/256), dim3(256), 0, stream>>>(targets, ws);
    ce_finalize_kernel<<<dim3(1), dim3(1), 0, stream>>>(ws, out);
}
